// CrossNonLocalBlock_74887049773276
// MI455X (gfx1250) — matrix operand order repaired and checked by reading, not compiled, not run on hardware
//
#include <hip/hip_runtime.h>

#define BATCH 16
#define CCH   256   // C
#define ICH   128   // IC
#define NSP   1024  // H*W

typedef __attribute__((ext_vector_type(16))) __bf16          v16bf;
typedef __attribute__((ext_vector_type(8)))  float           v8f;
typedef __attribute__((ext_vector_type(8)))  unsigned short  us8;
typedef __attribute__((ext_vector_type(16))) unsigned short  us16;

// ---------- bf16 helpers (RNE, bit-exact) ----------
static __device__ __forceinline__ unsigned short f32_to_bf16_bits(float f) {
  union { float f; unsigned int u; } v; v.f = f;
  unsigned int u = v.u;
  u += 0x7FFFu + ((u >> 16) & 1u);
  return (unsigned short)(u >> 16);
}
static __device__ __forceinline__ __bf16 bf16_from_bits(unsigned short s) {
  union { unsigned short u; __bf16 b; } v; v.u = s; return v.b;
}
static __device__ __forceinline__ void store_out(float* p, float v)          { *p = v; }
static __device__ __forceinline__ void store_out(unsigned short* p, float v) { *p = f32_to_bf16_bits(v); }

// ---------- vectorized fragment loads (pure bf16) ----------
// A fragment: lane holds 16 contiguous K values: [abase, abase+16)
static __device__ __forceinline__ v16bf load_a_frag(const unsigned short* row, int abase) {
  us8 c0 = *(const us8*)(row + abase);
  us8 c1 = *(const us8*)(row + abase + 8);
  v16bf a;
#pragma unroll
  for (int e = 0; e < 8; ++e) { a[e] = bf16_from_bits(c0[e]); a[e + 8] = bf16_from_bits(c1[e]); }
  return a;
}
// B fragment (from B^T storage): lane holds 16 contiguous K values at fixed column
static __device__ __forceinline__ v16bf load_b_frag(const unsigned short* row, int bbase) {
  us16 c = *(const us16*)(row + bbase);
  v16bf v;
#pragma unroll
  for (int e = 0; e < 16; ++e) v[e] = bf16_from_bits(c[e]);
  return v;
}

// ---------------------------------------------------------------------
// Batched WMMA GEMM, all operands bf16, contiguous-per-lane:
//   A(m,k) = A[m*lda + k]      (row-major)
//   B(k,n) = Bt[n*ldb + k]     (B stored transposed)
//   D(m,n) -> TRANS_D ? D[n*ldd+m] : D[m*ldd+n]   (f32 or bf16)
// Wave tile 32x32 (2x2 WMMA), block = 4 waves (2x2) -> 64x64 per block.
// ---------------------------------------------------------------------
template <typename TD, bool TRANS_D, bool RELU, bool HAS_BIAS, bool ACCUM, bool RESID>
__global__ void wmma_gemm_k(const unsigned short* __restrict__ A, long long bsA, int lda,
                            const unsigned short* __restrict__ Bt, long long bsB, int ldb,
                            TD* __restrict__ D, long long bsD, int ldd,
                            const float* __restrict__ bias,
                            const float* __restrict__ resid, long long bsR,
                            int K) {
  const int b    = blockIdx.z;
  const int wave = threadIdx.x >> 5;
  const int lane = threadIdx.x & 31;
  const int wm   = wave >> 1, wn = wave & 1;
  const int m0   = blockIdx.x * 64 + wm * 32;
  const int n0   = blockIdx.y * 64 + wn * 32;
  const int half = lane >> 4, l16 = lane & 15;

  const unsigned short* Ab = A  + (long long)b * bsA;
  const unsigned short* Bb = Bt + (long long)b * bsB;
  const unsigned short* arow0 = Ab + (long long)(m0 + l16) * lda;
  const unsigned short* arow1 = arow0 + (long long)16 * lda;
  const unsigned short* brow0 = Bb + (long long)(n0 + l16) * ldb;
  const unsigned short* brow1 = brow0 + (long long)16 * ldb;

  v8f acc00 = {}, acc01 = {}, acc10 = {}, acc11 = {};

  for (int kk = 0; kk < K; kk += 32) {
    const int abase = kk + (half ? 16 : 0);
    const int bbase = kk + (half ? 16 : 0);
    // prefetch 2 K-steps ahead (speculative; OOB dropped) -> global_prefetch_b8
    __builtin_prefetch((const void*)(arow0 + abase + 64), 0, 3);
    __builtin_prefetch((const void*)(brow0 + bbase + 64), 0, 3);
    v16bf a0 = load_a_frag(arow0, abase);
    v16bf a1 = load_a_frag(arow1, abase);
    v16bf b0 = load_b_frag(brow0, bbase);
    v16bf b1 = load_b_frag(brow1, bbase);
    acc00 = __builtin_amdgcn_wmma_f32_16x16x32_bf16(false, a0, false, b0, (short)0, acc00, false, false);
    acc01 = __builtin_amdgcn_wmma_f32_16x16x32_bf16(false, a0, false, b1, (short)0, acc01, false, false);
    acc10 = __builtin_amdgcn_wmma_f32_16x16x32_bf16(false, a1, false, b0, (short)0, acc10, false, false);
    acc11 = __builtin_amdgcn_wmma_f32_16x16x32_bf16(false, a1, false, b1, (short)0, acc11, false, false);
  }

  TD* Db = D + (long long)b * bsD;
  const float* Rb = RESID ? (resid + (long long)b * bsR) : nullptr;
  const v8f* accs[4] = { &acc00, &acc01, &acc10, &acc11 };
#pragma unroll
  for (int t = 0; t < 4; ++t) {
    const int ti = t >> 1, tj = t & 1;
#pragma unroll
    for (int r = 0; r < 8; ++r) {
      const int m = m0 + ti * 16 + (half ? 8 : 0) + r;
      const int n = n0 + tj * 16 + l16;
      float v = (*accs[t])[r];
      if (HAS_BIAS) v += bias[m];
      if (RELU)     v = v > 0.f ? v : 0.f;
      const long long idx = TRANS_D ? ((long long)n * ldd + m) : ((long long)m * ldd + n);
      if (RESID) v += Rb[idx];
      if (ACCUM) v += (float)Db[idx];
      store_out(&Db[idx], v);
    }
  }
}

// ---------------------------------------------------------------------
// f32 -> bf16 elementwise (one-shot weight conversion)
__global__ void cvt_bf16_k(const float* __restrict__ in, unsigned short* __restrict__ out,
                           int total) {
  const int i = blockIdx.x * blockDim.x + threadIdx.x;
  if (i >= total) return;
  out[i] = f32_to_bf16_bits(in[i]);
}

// [B,C,N] f32 -> [B,N,C] bf16 (coalesced writes)
__global__ void transpose_cn_bf_k(const float* __restrict__ in, unsigned short* __restrict__ out,
                                  int C, int N, int total) {
  const int i = blockIdx.x * blockDim.x + threadIdx.x;
  if (i >= total) return;
  const int c = i % C;
  const int n = (i / C) % N;
  const int b = i / (C * N);
  out[i] = f32_to_bf16_bits(in[((long long)b * C + c) * N + n]);
}

// d[n] = 0.5*(rowsum + colsum); dinv = d!=0 ? rsqrt(d) : 0
__global__ void degree_k(const float* __restrict__ att, float* __restrict__ dinv, int N) {
  const int b = blockIdx.y;
  const int n = blockIdx.x * blockDim.x + threadIdx.x;
  if (n >= N) return;
  const float* A = att + (long long)b * N * N;
  float s = 0.f;
  for (int m = 0; m < N; ++m)
    s += A[(long long)n * N + m] + A[(long long)m * N + n];
  const float d = 0.5f * s;
  dinv[b * N + n] = (d != 0.f) ? rsqrtf(d) : 0.f;
}

// f[n,m] = 0.5*(att[n,m]+att[m,n]) * dinv[n]*dinv[m]  -> bf16
__global__ void symnorm_k(const float* __restrict__ att, const float* __restrict__ dinv,
                          unsigned short* __restrict__ f, int N) {
  const int b = blockIdx.z;
  const int n = blockIdx.y;
  const int m = blockIdx.x * blockDim.x + threadIdx.x;
  if (m >= N) return;
  const long long base = (long long)b * N * N;
  const float v = 0.5f * (att[base + (long long)n * N + m] + att[base + (long long)m * N + n])
                  * dinv[b * N + n] * dinv[b * N + m];
  f[base + (long long)n * N + m] = f32_to_bf16_bits(v);
}

// per-channel batch stats (mean, rsqrt(var+eps)) for both BN inputs
__global__ void bn_stats_k(const float* __restrict__ s1, const float* __restrict__ s2,
                           float* __restrict__ stats, int Bc, int C, int N) {
  const int c = blockIdx.x;
  const float* src = (blockIdx.y == 0) ? s1 : s2;
  const int tid = threadIdx.x;
  float sum = 0.f, sq = 0.f;
  const int total = Bc * N;
  for (int i = tid; i < total; i += blockDim.x) {
    const int b = i / N, n = i - b * N;
    const float v = src[(long long)b * C * N + (long long)c * N + n];
    sum += v; sq += v * v;
  }
  __shared__ float ssum[256], ssq[256];
  ssum[tid] = sum; ssq[tid] = sq;
  __syncthreads();
  for (int s = 128; s > 0; s >>= 1) {
    if (tid < s) { ssum[tid] += ssum[tid + s]; ssq[tid] += ssq[tid + s]; }
    __syncthreads();
  }
  if (tid == 0) {
    const float inv = 1.f / (float)total;
    const float mu  = ssum[0] * inv;
    const float var = ssq[0] * inv - mu * mu;
    stats[(blockIdx.y * C + c) * 2 + 0] = mu;
    stats[(blockIdx.y * C + c) * 2 + 1] = rsqrtf(var + 1e-5f);
  }
}

// zT[b,n,c] = bn1(s1)[b,c,n] + bn2(s2)[b,c,n] + xs[b,c,n]  -> bf16 (coalesced writes)
__global__ void bn_combine_t_k(const float* __restrict__ s1, const float* __restrict__ s2,
                               const float* __restrict__ xs, unsigned short* __restrict__ zT,
                               const float* __restrict__ stats,
                               const float* __restrict__ g1, const float* __restrict__ b1,
                               const float* __restrict__ g2, const float* __restrict__ b2,
                               int C, int N, int total) {
  const int i = blockIdx.x * blockDim.x + threadIdx.x;
  if (i >= total) return;
  const int c = i % C;
  const int n = (i / C) % N;
  const int b = i / (C * N);
  const long long src = ((long long)b * C + c) * N + n;
  const float mu1 = stats[c * 2 + 0],       rs1 = stats[c * 2 + 1];
  const float mu2 = stats[(C + c) * 2 + 0], rs2 = stats[(C + c) * 2 + 1];
  const float v1 = (s1[src] - mu1) * rs1 * g1[c] + b1[c];
  const float v2 = (s2[src] - mu2) * rs2 * g2[c] + b2[c];
  zT[i] = f32_to_bf16_bits(v1 + v2 + xs[src]);
}

// ---------------------------------------------------------------------
static void run_agg_conv(const unsigned short* f, const unsigned short* g,
                         const unsigned short* Wbf, const float* bias, float* dst, bool accum,
                         unsigned short* aggbuf, hipStream_t stream) {
  const dim3 blk(128);
  // agg[b,n,i] = sum_m f[n,m] * g_tokens[m,i];  g stored [IC,N] -> B^T access
  wmma_gemm_k<unsigned short, false, false, false, false, false>
      <<<dim3(NSP / 64, ICH / 64, BATCH), blk, 0, stream>>>(
          f, (long long)NSP * NSP, NSP,
          g, (long long)ICH * NSP, NSP,
          aggbuf, (long long)NSP * ICH, ICH,
          nullptr, nullptr, 0, NSP);
  // dst[b,c,n] (+)= W[c,:] . agg[b,n,:] + bias[c];  agg [N,IC] -> B^T access
  if (accum) {
    wmma_gemm_k<float, false, false, true, true, false>
        <<<dim3(CCH / 64, NSP / 64, BATCH), blk, 0, stream>>>(
            Wbf, 0, ICH, aggbuf, (long long)NSP * ICH, ICH,
            dst, (long long)CCH * NSP, NSP, bias, nullptr, 0, ICH);
  } else {
    wmma_gemm_k<float, false, false, true, false, false>
        <<<dim3(CCH / 64, NSP / 64, BATCH), blk, 0, stream>>>(
            Wbf, 0, ICH, aggbuf, (long long)NSP * ICH, ICH,
            dst, (long long)CCH * NSP, NSP, bias, nullptr, 0, ICH);
  }
}

extern "C" void kernel_launch(void* const* d_in, const int* in_sizes, int n_in,
                              void* d_out, int out_size, void* d_ws, size_t ws_size,
                              hipStream_t stream) {
  (void)in_sizes; (void)n_in; (void)out_size; (void)ws_size;

  const float* x    = (const float*)d_in[0];
  const float* ob   = (const float*)d_in[1];
  const float* od   = (const float*)d_in[2];
  const float* Wt   = (const float*)d_in[3];
  const float* Wp   = (const float*)d_in[4];
  const float* Wg_[3] = { (const float*)d_in[5], (const float*)d_in[7], (const float*)d_in[9] };
  const float* bg_[3] = { (const float*)d_in[6], (const float*)d_in[8], (const float*)d_in[10] };
  const float* Wwx  = (const float*)d_in[11]; const float* bwx  = (const float*)d_in[12];
  const float* Wwb  = (const float*)d_in[13]; const float* bwb  = (const float*)d_in[14];
  const float* Wwd  = (const float*)d_in[15]; const float* bwd  = (const float*)d_in[16];
  const float* Wwxb = (const float*)d_in[17]; const float* bwxb = (const float*)d_in[18];
  const float* Wwxd = (const float*)d_in[19]; const float* bwxd = (const float*)d_in[20];
  const float* gamma1 = (const float*)d_in[21]; const float* beta1 = (const float*)d_in[22];
  const float* gamma2 = (const float*)d_in[23]; const float* beta2 = (const float*)d_in[24];
  const float* Wout = (const float*)d_in[25]; const float* bout = (const float*)d_in[26];
  float* out = (float*)d_out;

  // ---- carve workspace ----
  char* w = (char*)d_ws;
  auto carve = [&](size_t bytes) -> void* {
    void* p = (void*)w;
    w += (bytes + 255) & ~(size_t)255;
    return p;
  };
  unsigned short* tT  = (unsigned short*)carve((size_t)BATCH * NSP * ICH * 2); // [B,N,IC]
  unsigned short* pT  = (unsigned short*)carve((size_t)BATCH * NSP * ICH * 2); // [B,N,IC]
  float*          att = (float*)carve((size_t)BATCH * NSP * NSP * 4);
  unsigned short* fb[3];
  for (int i = 0; i < 3; ++i) fb[i] = (unsigned short*)carve((size_t)BATCH * NSP * NSP * 2);
  unsigned short* gb[3];                                                        // [B,IC,N]
  for (int i = 0; i < 3; ++i) gb[i] = (unsigned short*)carve((size_t)BATCH * ICH * NSP * 2);
  unsigned short* aggbuf = (unsigned short*)carve((size_t)BATCH * NSP * ICH * 2); // [B,N,IC]
  float* sum1 = (float*)carve((size_t)BATCH * CCH * NSP * 4);
  float* sum2 = (float*)carve((size_t)BATCH * CCH * NSP * 4);
  float* bufx = (float*)carve((size_t)BATCH * CCH * NSP * 4);
  unsigned short* xT = (unsigned short*)carve((size_t)BATCH * NSP * CCH * 2); // [B,N,C] bf16; reused as zT
  float* dinv  = (float*)carve((size_t)BATCH * NSP * 4);
  float* stats = (float*)carve((size_t)2 * CCH * 2 * 4);
  // one-shot bf16 weight copies
  unsigned short* WtB   = (unsigned short*)carve((size_t)ICH * CCH * 2);
  unsigned short* WpB   = (unsigned short*)carve((size_t)ICH * CCH * 2);
  unsigned short* WgB[3];
  for (int i = 0; i < 3; ++i) WgB[i] = (unsigned short*)carve((size_t)ICH * CCH * 2);
  unsigned short* WwxB  = (unsigned short*)carve((size_t)CCH * ICH * 2);
  unsigned short* WwbB  = (unsigned short*)carve((size_t)CCH * ICH * 2);
  unsigned short* WwdB  = (unsigned short*)carve((size_t)CCH * ICH * 2);
  unsigned short* WwxbB = (unsigned short*)carve((size_t)CCH * ICH * 2);
  unsigned short* WwxdB = (unsigned short*)carve((size_t)CCH * ICH * 2);
  unsigned short* WoutB = (unsigned short*)carve((size_t)CCH * CCH * 2);

  const dim3 blk(128);
  const int totalBCN = BATCH * CCH * NSP;
  const float* inps[3] = { x, ob, od };

  // ---- one-shot f32 -> bf16 weight conversion ----
  {
    const int wsz = ICH * CCH;                      // 32768
    const dim3 cb(256), cg((wsz + 255) / 256);
    cvt_bf16_k<<<cg, cb, 0, stream>>>(Wt, WtB, wsz);
    cvt_bf16_k<<<cg, cb, 0, stream>>>(Wp, WpB, wsz);
    for (int i = 0; i < 3; ++i) cvt_bf16_k<<<cg, cb, 0, stream>>>(Wg_[i], WgB[i], wsz);
    cvt_bf16_k<<<cg, cb, 0, stream>>>(Wwx,  WwxB,  wsz);
    cvt_bf16_k<<<cg, cb, 0, stream>>>(Wwb,  WwbB,  wsz);
    cvt_bf16_k<<<cg, cb, 0, stream>>>(Wwd,  WwdB,  wsz);
    cvt_bf16_k<<<cg, cb, 0, stream>>>(Wwxb, WwxbB, wsz);
    cvt_bf16_k<<<cg, cb, 0, stream>>>(Wwxd, WwxdB, wsz);
    const int osz = CCH * CCH;                      // 65536
    cvt_bf16_k<<<dim3((osz + 255) / 256), cb, 0, stream>>>(Wout, WoutB, osz);
  }

  for (int i = 0; i < 3; ++i) {
    // xT[b,n,c] = bf16(X[b,c,n])
    transpose_cn_bf_k<<<dim3((totalBCN + 255) / 256), dim3(256), 0, stream>>>(
        inps[i], xT, CCH, NSP, totalBCN);
    // tT[b,n,:] = relu(Wt @ X)^T   (TRANS_D store, B^T = xT)
    wmma_gemm_k<unsigned short, true, true, false, false, false>
        <<<dim3(ICH / 64, NSP / 64, BATCH), blk, 0, stream>>>(
            WtB, 0, CCH, xT, (long long)NSP * CCH, CCH,
            tT, (long long)NSP * ICH, ICH, nullptr, nullptr, 0, CCH);
    // pT
    wmma_gemm_k<unsigned short, true, true, false, false, false>
        <<<dim3(ICH / 64, NSP / 64, BATCH), blk, 0, stream>>>(
            WpB, 0, CCH, xT, (long long)NSP * CCH, CCH,
            pT, (long long)NSP * ICH, ICH, nullptr, nullptr, 0, CCH);
    // att[n,m] = tT[n,:] . pT[m,:]   (A natural, B^T natural)
    wmma_gemm_k<float, false, false, false, false, false>
        <<<dim3(NSP / 64, NSP / 64, BATCH), blk, 0, stream>>>(
            tT, (long long)NSP * ICH, ICH, pT, (long long)NSP * ICH, ICH,
            att, (long long)NSP * NSP, NSP, nullptr, nullptr, 0, ICH);
    degree_k<<<dim3(NSP / 256, BATCH), dim3(256), 0, stream>>>(att, dinv, NSP);
    symnorm_k<<<dim3(NSP / 256, NSP, BATCH), dim3(256), 0, stream>>>(att, dinv, fb[i], NSP);
    // g[b,i,n] = Wg @ X + bg  stored [B,IC,N] (natural D; agg reads it as B^T)
    wmma_gemm_k<unsigned short, false, false, true, false, false>
        <<<dim3(ICH / 64, NSP / 64, BATCH), blk, 0, stream>>>(
            WgB[i], 0, CCH, xT, (long long)NSP * CCH, CCH,
            gb[i], (long long)ICH * NSP, NSP, bg_[i], nullptr, 0, CCH);
  }

  run_agg_conv(fb[2], gb[2], WwdB,  bwd,  sum1, false, aggbuf, stream); // od_self
  run_agg_conv(fb[1], gb[0], WwxbB, bwxb, sum1, true,  aggbuf, stream); // + x_ob_cross
  run_agg_conv(fb[1], gb[1], WwbB,  bwb,  sum2, false, aggbuf, stream); // ob_self
  run_agg_conv(fb[2], gb[0], WwxdB, bwxd, sum2, true,  aggbuf, stream); // + x_od_cross
  run_agg_conv(fb[0], gb[0], WwxB,  bwx,  bufx, false, aggbuf, stream); // x_self

  bn_stats_k<<<dim3(CCH, 2), dim3(256), 0, stream>>>(sum1, sum2, stats, BATCH, CCH, NSP);
  // zT[b,n,c] = bf16( bn1(sum1) + bn2(sum2) + x_self )   (xT buffer reused as zT)
  bn_combine_t_k<<<dim3((totalBCN + 255) / 256), dim3(256), 0, stream>>>(
      sum1, sum2, bufx, xT, stats, gamma1, beta1, gamma2, beta2, CCH, NSP, totalBCN);

  // out = Wout @ z + bout + x   (B^T = zT, residual x)
  wmma_gemm_k<float, false, false, true, false, true>
      <<<dim3(CCH / 64, NSP / 64, BATCH), blk, 0, stream>>>(
          WoutB, 0, CCH, xT, (long long)NSP * CCH, CCH,
          out, (long long)CCH * NSP, NSP, bout, x, (long long)CCH * NSP, CCH);
}
